// TransformerBlock_74371653697644
// MI455X (gfx1250) — matrix operand order repaired and checked by reading, not compiled, not run on hardware
//
#include <hip/hip_runtime.h>
#include <stdint.h>

typedef unsigned short bf16_t;
typedef __attribute__((ext_vector_type(16))) __bf16 v16bf;
typedef __attribute__((ext_vector_type(8)))  float  v8f;

union FragU { uint32_t u[8]; v16bf v; };

__device__ inline bf16_t f2bf(float f) {
  uint32_t u = __float_as_uint(f);
  u += 0x7fffu + ((u >> 16) & 1u);
  return (bf16_t)(u >> 16);
}
__device__ inline float bf2f(bf16_t h) { return __uint_as_float(((uint32_t)h) << 16); }
__device__ inline float fast_rcp(float x) { return __builtin_amdgcn_rcpf(x); }

// Async global->LDS copy (16B per lane), tracked by ASYNCcnt.  [ISA 10, §15.18.3]
__device__ inline void async_b128(uint32_t lds_off, const void* gaddr) {
  asm volatile("global_load_async_to_lds_b128 %0, %1, off"
               :: "v"(lds_off), "v"((unsigned long long)gaddr) : "memory");
}
// Async loads complete in order -> waiting ASYNCcnt<=N (N = ops in flight for the
// prefetched tile) guarantees the current tile is fully resident.
template <int N>
__device__ inline void wait_async() {
  asm volatile("s_wait_asynccnt %0" :: "i"(N) : "memory");
}
__device__ inline uint32_t lds_addr(const void* p) { return (uint32_t)(uintptr_t)p; }

// A fragment (16x32 bf16, MxK): lane m=lane&15, half=lane>>4.
// VGPR j holds K = (j&3)*2 + (j>>2)*16 + half*8 (+0,+1)   [ISA 7.12.2]
__device__ inline v16bf load_a_frag(const bf16_t* lds, int stride, int mbase, int kbase) {
  int lane = threadIdx.x & 31;
  int half = lane >> 4, m = lane & 15;
  FragU f;
#pragma unroll
  for (int j = 0; j < 8; ++j) {
    int k2 = (j & 3) * 2 + (j >> 2) * 16 + half * 8;
    f.u[j] = *(const uint32_t*)&lds[(mbase + m) * stride + kbase + k2];
  }
  return f.v;
}
// B fragment (32x16 bf16, KxN), lds laid out [n][k]
__device__ inline v16bf load_b_frag(const bf16_t* lds, int stride, int nbase, int kbase) {
  int lane = threadIdx.x & 31;
  int half = lane >> 4, n = lane & 15;
  FragU f;
#pragma unroll
  for (int j = 0; j < 8; ++j) {
    int k2 = (j & 3) * 2 + (j >> 2) * 16 + half * 8;
    f.u[j] = *(const uint32_t*)&lds[(nbase + n) * stride + kbase + k2];
  }
  return f.v;
}
__device__ inline v8f wmma_bf16(v16bf a, v16bf b, v8f c) {
  return __builtin_amdgcn_wmma_f32_16x16x32_bf16(false, a, false, b, (short)0, c, false, false);
}

// ---------------- fp32 -> bf16 conversion ----------------
__global__ __launch_bounds__(256) void cvt_bf16_kernel(const float* __restrict__ in,
                                                       bf16_t* __restrict__ out, int n) {
  int i = blockIdx.x * 256 + threadIdx.x;
  if (i < n) out[i] = f2bf(in[i]);
}

// ---------------- RMSNorm: wave per token, out bf16 ----------------
__global__ __launch_bounds__(256) void rmsnorm_kernel(const float* __restrict__ x,
                                                      const float* __restrict__ w,
                                                      bf16_t* __restrict__ out) {
  int wid = threadIdx.x >> 5, lane = threadIdx.x & 31;
  int t = blockIdx.x * 8 + wid;
  const float* xp = x + (size_t)t * 1024 + lane * 32;
  float vals[32];
  float ss = 0.f;
#pragma unroll
  for (int i = 0; i < 8; ++i) {
    float4 v = *(const float4*)(xp + i * 4);
    vals[i*4+0] = v.x; vals[i*4+1] = v.y; vals[i*4+2] = v.z; vals[i*4+3] = v.w;
    ss += v.x*v.x + v.y*v.y + v.z*v.z + v.w*v.w;
  }
#pragma unroll
  for (int off = 16; off > 0; off >>= 1) ss += __shfl_xor(ss, off, 32);
  float rr = rsqrtf(ss * (1.f / 1024.f) + 1e-5f);
  bf16_t* op = out + (size_t)t * 1024 + lane * 32;
  const float* wp = w + lane * 32;
#pragma unroll
  for (int i = 0; i < 32; i += 2) {
    uint32_t p = (uint32_t)f2bf(vals[i] * rr * wp[i]) |
                 ((uint32_t)f2bf(vals[i+1] * rr * wp[i+1]) << 16);
    *(uint32_t*)(op + i) = p;
  }
}

// ---------------- RoPE: fp32 in -> bf16 out ----------------
__global__ __launch_bounds__(256) void rope_kernel(const float* __restrict__ in,
                                                   bf16_t* __restrict__ out, int total) {
  int idx = blockIdx.x * 256 + threadIdx.x;
  if (idx >= total) return;
  int pair = idx & 31;
  int h = (idx >> 5) & 15;
  int tok = idx >> 9;
  int pos = tok & 1023;  // L = 1024
  float inv = __powf(10000.f, -(float)pair * (1.f / 32.f));
  float s, c;
  __sincosf((float)pos * inv, &s, &c);
  size_t base = (size_t)tok * 1024 + h * 64;
  float x1 = in[base + pair], x2 = in[base + 32 + pair];
  out[base + pair]      = f2bf(x1 * c - x2 * s);
  out[base + 32 + pair] = f2bf(x1 * s + x2 * c);
}

// ---------------- generic bf16 WMMA GEMM:  C[M,N] = A[M,K] @ W[N,K]^T ----------
// Double-buffered async-to-LDS pipeline: prefetch tile s+1 while computing s.
// EPI: 0 = f32 store, 1 = bf16 store, 2 = f32 store with residual add
template <int EPI>
__global__ __launch_bounds__(256)
void gemm_bf16_kernel(const bf16_t* __restrict__ A, const bf16_t* __restrict__ W,
                      const float* __restrict__ R, float* __restrict__ Cf,
                      bf16_t* __restrict__ Cb, int N, int K) {
  __shared__ __align__(16) bf16_t sA[2][128 * 64];
  __shared__ __align__(16) bf16_t sB[2][64 * 64];
  int tid = threadIdx.x;
  int wid = tid >> 5;
  int wm = (wid & 3) * 32, wn = (wid >> 2) * 32;
  int rowBase = blockIdx.y * 128, colBase = blockIdx.x * 64;
  v8f acc[2][2] = {};

  auto stage = [&](int buf, int kb) {
#pragma unroll
    for (int i = 0; i < 4; ++i) {
      int idx = tid + i * 256;
      int r = idx >> 3, c = (idx & 7) * 8;
      async_b128(lds_addr(&sA[buf][r * 64 + c]), &A[(size_t)(rowBase + r) * K + kb + c]);
    }
#pragma unroll
    for (int i = 0; i < 2; ++i) {
      int idx = tid + i * 256;
      int r = idx >> 3, c = (idx & 7) * 8;
      async_b128(lds_addr(&sB[buf][r * 64 + c]), &W[(size_t)(colBase + r) * K + kb + c]);
    }
  };

  int nsteps = K >> 6;
  stage(0, 0);
  for (int s = 0; s < nsteps; ++s) {
    int cur = s & 1;
    if (s + 1 < nsteps) { stage(cur ^ 1, (s + 1) * 64); wait_async<6>(); }
    else wait_async<0>();
    __syncthreads();
#pragma unroll
    for (int ks = 0; ks < 2; ++ks) {
      v16bf af[2], bfr[2];
#pragma unroll
      for (int mi = 0; mi < 2; ++mi) af[mi] = load_a_frag(sA[cur], 64, wm + mi * 16, ks * 32);
#pragma unroll
      for (int ni = 0; ni < 2; ++ni) bfr[ni] = load_b_frag(sB[cur], 64, wn + ni * 16, ks * 32);
#pragma unroll
      for (int mi = 0; mi < 2; ++mi)
#pragma unroll
        for (int ni = 0; ni < 2; ++ni)
          acc[mi][ni] = wmma_bf16(af[mi], bfr[ni], acc[mi][ni]);
    }
    __syncthreads();  // all waves done reading `cur` before it is re-staged
  }
  int lane = tid & 31, half = lane >> 4, nlo = lane & 15;
#pragma unroll
  for (int mi = 0; mi < 2; ++mi)
#pragma unroll
    for (int ni = 0; ni < 2; ++ni)
#pragma unroll
      for (int r = 0; r < 8; ++r) {
        int row = rowBase + wm + mi * 16 + half * 8 + r;
        int col = colBase + wn + ni * 16 + nlo;
        float v = acc[mi][ni][r];
        if (EPI == 0) Cf[(size_t)row * N + col] = v;
        else if (EPI == 1) Cb[(size_t)row * N + col] = f2bf(v);
        else Cf[(size_t)row * N + col] = v + R[(size_t)row * N + col];
      }
}

// ---------------- SwiGLU GEMM:  out = silu(A@W1^T) * (A@W2^T), bf16 out ------
__global__ __launch_bounds__(256)
void gemm_silumul_kernel(const bf16_t* __restrict__ A, const bf16_t* __restrict__ W1,
                         const bf16_t* __restrict__ W2, bf16_t* __restrict__ Cb,
                         int N, int K) {
  __shared__ __align__(16) bf16_t sA[2][128 * 64];
  __shared__ __align__(16) bf16_t sB1[2][64 * 64];
  __shared__ __align__(16) bf16_t sB2[2][64 * 64];
  int tid = threadIdx.x;
  int wid = tid >> 5;
  int wm = (wid & 3) * 32, wn = (wid >> 2) * 32;
  int rowBase = blockIdx.y * 128, colBase = blockIdx.x * 64;
  v8f acc1[2][2] = {}, acc2[2][2] = {};

  auto stage = [&](int buf, int kb) {
#pragma unroll
    for (int i = 0; i < 4; ++i) {
      int idx = tid + i * 256;
      int r = idx >> 3, c = (idx & 7) * 8;
      async_b128(lds_addr(&sA[buf][r * 64 + c]), &A[(size_t)(rowBase + r) * K + kb + c]);
    }
#pragma unroll
    for (int i = 0; i < 2; ++i) {
      int idx = tid + i * 256;
      int r = idx >> 3, c = (idx & 7) * 8;
      async_b128(lds_addr(&sB1[buf][r * 64 + c]), &W1[(size_t)(colBase + r) * K + kb + c]);
      async_b128(lds_addr(&sB2[buf][r * 64 + c]), &W2[(size_t)(colBase + r) * K + kb + c]);
    }
  };

  int nsteps = K >> 6;
  stage(0, 0);
  for (int s = 0; s < nsteps; ++s) {
    int cur = s & 1;
    if (s + 1 < nsteps) { stage(cur ^ 1, (s + 1) * 64); wait_async<8>(); }
    else wait_async<0>();
    __syncthreads();
#pragma unroll
    for (int ks = 0; ks < 2; ++ks) {
      v16bf af[2], b1[2], b2[2];
#pragma unroll
      for (int mi = 0; mi < 2; ++mi) af[mi] = load_a_frag(sA[cur], 64, wm + mi * 16, ks * 32);
#pragma unroll
      for (int ni = 0; ni < 2; ++ni) {
        b1[ni] = load_b_frag(sB1[cur], 64, wn + ni * 16, ks * 32);
        b2[ni] = load_b_frag(sB2[cur], 64, wn + ni * 16, ks * 32);
      }
#pragma unroll
      for (int mi = 0; mi < 2; ++mi)
#pragma unroll
        for (int ni = 0; ni < 2; ++ni) {
          acc1[mi][ni] = wmma_bf16(af[mi], b1[ni], acc1[mi][ni]);
          acc2[mi][ni] = wmma_bf16(af[mi], b2[ni], acc2[mi][ni]);
        }
    }
    __syncthreads();
  }
  int lane = tid & 31, half = lane >> 4, nlo = lane & 15;
#pragma unroll
  for (int mi = 0; mi < 2; ++mi)
#pragma unroll
    for (int ni = 0; ni < 2; ++ni)
#pragma unroll
      for (int r = 0; r < 8; ++r) {
        int row = rowBase + wm + mi * 16 + half * 8 + r;
        int col = colBase + wn + ni * 16 + nlo;
        float a = acc1[mi][ni][r];
        float g = a * fast_rcp(1.f + __expf(-a)) * acc2[mi][ni][r];
        Cb[(size_t)row * N + col] = f2bf(g);
      }
}

// ---------------- flash attention (non-causal, zero mask) ----------------
// grid: (L/64, B*HEADS), block 128 (4 waves, 16 query rows each)
__global__ __launch_bounds__(128)
void attn_kernel(const bf16_t* __restrict__ Q, const bf16_t* __restrict__ K,
                 const bf16_t* __restrict__ V, bf16_t* __restrict__ O) {
  __shared__ __align__(16) bf16_t sQ[64 * 64];
  __shared__ __align__(16) bf16_t sK[32 * 64];
  __shared__ __align__(16) bf16_t sVt[64 * 32];  // [d][key]
  __shared__ __align__(16) bf16_t sP[4 * 16 * 32];
  int tid = threadIdx.x;
  int bh = blockIdx.y, b = bh >> 4, h = bh & 15;
  int q0 = blockIdx.x * 64;
#pragma unroll
  for (int i = 0; i < 4; ++i) {
    int idx = tid + i * 128;
    int r = idx >> 3, c = (idx & 7) * 8;
    *(uint4*)&sQ[r * 64 + c] = *(const uint4*)&Q[((size_t)(b * 1024 + q0 + r)) * 1024 + h * 64 + c];
  }
  __syncthreads();
  int wid = tid >> 5, lane = tid & 31, half = lane >> 4, nlo = lane & 15;
  int qr = wid * 16;
  v16bf qf[2];
  qf[0] = load_a_frag(sQ, 64, qr, 0);
  qf[1] = load_a_frag(sQ, 64, qr, 32);
  v8f accO[4] = {};
  float mrow[8], srow[8];
#pragma unroll
  for (int r = 0; r < 8; ++r) { mrow[r] = -INFINITY; srow[r] = 0.f; }

  for (int kb = 0; kb < 1024; kb += 32) {
    __syncthreads();
#pragma unroll
    for (int i = 0; i < 2; ++i) {
      int idx = tid + i * 128;
      int r = idx >> 3, c = (idx & 7) * 8;
      // K tile through the async path; V needs a transpose scatter (manual).
      async_b128(lds_addr(&sK[r * 64 + c]),
                 &K[((size_t)(b * 1024 + kb + r)) * 1024 + h * 64 + c]);
      uint4 d = *(const uint4*)&V[((size_t)(b * 1024 + kb + r)) * 1024 + h * 64 + c];
      const bf16_t* dv = (const bf16_t*)&d;
#pragma unroll
      for (int j = 0; j < 8; ++j) sVt[(c + j) * 32 + r] = dv[j];
    }
    wait_async<0>();
    __syncthreads();
    v8f sacc[2] = {};
#pragma unroll
    for (int kc = 0; kc < 2; ++kc)
#pragma unroll
      for (int ks = 0; ks < 2; ++ks) {
        v16bf bk = load_b_frag(sK, 64, kc * 16, ks * 32);
        sacc[kc] = wmma_bf16(qf[ks], bk, sacc[kc]);
      }
    float p0[8], p1[8], sc[8];
#pragma unroll
    for (int r = 0; r < 8; ++r) {
      float s0 = sacc[0][r] * 0.125f, s1 = sacc[1][r] * 0.125f;
      float m = fmaxf(s0, s1);
#pragma unroll
      for (int off = 1; off < 16; off <<= 1) m = fmaxf(m, __shfl_xor(m, off, 16));
      float mn = fmaxf(mrow[r], m);
      sc[r] = __expf(mrow[r] - mn);
      p0[r] = __expf(s0 - mn);
      p1[r] = __expf(s1 - mn);
      float rsum = p0[r] + p1[r];
#pragma unroll
      for (int off = 1; off < 16; off <<= 1) rsum += __shfl_xor(rsum, off, 16);
      srow[r] = srow[r] * sc[r] + rsum;
      mrow[r] = mn;
    }
#pragma unroll
    for (int t4 = 0; t4 < 4; ++t4)
#pragma unroll
      for (int r = 0; r < 8; ++r) accO[t4][r] *= sc[r];
    bf16_t* pp = &sP[wid * 512];
#pragma unroll
    for (int r = 0; r < 8; ++r) {
      int row = half * 8 + r;
      pp[row * 32 + nlo]      = f2bf(p0[r]);
      pp[row * 32 + 16 + nlo] = f2bf(p1[r]);
    }
    asm volatile("s_wait_dscnt 0" ::: "memory");  // same-wave LDS RAW (P transpose)
    v16bf pf = load_a_frag(pp, 32, 0, 0);
#pragma unroll
    for (int t4 = 0; t4 < 4; ++t4) {
      v16bf bv = load_b_frag(sVt, 32, t4 * 16, 0);
      accO[t4] = wmma_bf16(pf, bv, accO[t4]);
    }
  }
  float rinv[8];
#pragma unroll
  for (int r = 0; r < 8; ++r) rinv[r] = fast_rcp(srow[r]);
#pragma unroll
  for (int t4 = 0; t4 < 4; ++t4)
#pragma unroll
    for (int r = 0; r < 8; ++r) {
      int row = q0 + qr + half * 8 + r;
      int col = h * 64 + t4 * 16 + nlo;
      O[((size_t)(b * 1024 + row)) * 1024 + col] = f2bf(accO[t4][r] * rinv[r]);
    }
}

// ---------------- gate: logits, top-2 softmax, expert list build -------------
__global__ __launch_bounds__(32) void zero_counts_kernel(int* counts) {
  if (threadIdx.x < 16) counts[threadIdx.x] = 0;
}

__global__ __launch_bounds__(256)
void gate_topk_kernel(const bf16_t* __restrict__ hn2, const float* __restrict__ gw,
                      int* __restrict__ counts, int* __restrict__ elist,
                      float* __restrict__ ewt) {
  int wid = threadIdx.x >> 5, lane = threadIdx.x & 31;
  int t = blockIdx.x * 8 + wid;
  float acc[16];
#pragma unroll
  for (int e = 0; e < 16; ++e) acc[e] = 0.f;
  for (int i = 0; i < 32; ++i) {
    int k = i * 32 + lane;
    float xv = bf2f(hn2[(size_t)t * 1024 + k]);
#pragma unroll
    for (int e = 0; e < 16; ++e) acc[e] += xv * gw[e * 1024 + k];
  }
#pragma unroll
  for (int e = 0; e < 16; ++e)
#pragma unroll
    for (int off = 16; off > 0; off >>= 1) acc[e] += __shfl_xor(acc[e], off, 32);
  if (lane == 0) {
    float v0 = -INFINITY; int i0 = 0;
#pragma unroll
    for (int e = 0; e < 16; ++e)
      if (acc[e] > v0) { v0 = acc[e]; i0 = e; }
    float v1 = -INFINITY; int i1 = 0;
#pragma unroll
    for (int e = 0; e < 16; ++e)
      if (e != i0 && acc[e] > v1) { v1 = acc[e]; i1 = e; }
    float e1 = __expf(v1 - v0);
    float inv = fast_rcp(1.f + e1);
    float w0 = inv, w1 = e1 * inv;
    int s0 = atomicAdd(&counts[i0], 1);
    elist[i0 * 4096 + s0] = t; ewt[i0 * 4096 + s0] = w0;
    int s1 = atomicAdd(&counts[i1], 1);
    elist[i1 * 4096 + s1] = t; ewt[i1 * 4096 + s1] = w1;
  }
}

// ---------------- fused MoE grouped GEMM (fc1+silu in LDS, fc2 scatter-add) --
// grid: (32 tiles, 16 experts), block 256
__global__ __launch_bounds__(256)
void moe_kernel(const bf16_t* __restrict__ hn2, const bf16_t* __restrict__ fc1,
                const bf16_t* __restrict__ fc2, const int* __restrict__ counts,
                const int* __restrict__ elist, const float* __restrict__ ewt,
                float* __restrict__ out) {
  __shared__ __align__(16) bf16_t sH[128 * 512];  // 128 KB intermediate (320 KB/WGP)
  __shared__ __align__(16) bf16_t sA[2][128 * 64];
  __shared__ __align__(16) bf16_t sB[2][64 * 64];
  __shared__ int   sTok[128];
  __shared__ float sWgt[128];
  int e = blockIdx.y, tile = blockIdx.x;
  int cnt = counts[e];
  if (tile * 128 >= cnt) return;
  int tid = threadIdx.x;
  if (tid < 128) {
    int slot = tile * 128 + tid;
    bool v = slot < cnt;
    sTok[tid] = v ? elist[e * 4096 + slot] : 0;
    sWgt[tid] = v ? ewt[e * 4096 + slot] : 0.f;
  }
  __syncthreads();
  int wid = tid >> 5, wm = (wid & 3) * 32, wn = (wid >> 2) * 32;
  int lane = tid & 31, half = lane >> 4, nlo = lane & 15;

  auto stage1 = [&](int buf, int nb, int kb) {
#pragma unroll
    for (int i = 0; i < 4; ++i) {
      int idx = tid + i * 256;
      int r = idx >> 3, c = (idx & 7) * 8;
      async_b128(lds_addr(&sA[buf][r * 64 + c]), &hn2[(size_t)sTok[r] * 1024 + kb + c]);
    }
#pragma unroll
    for (int i = 0; i < 2; ++i) {
      int idx = tid + i * 256;
      int r = idx >> 3, c = (idx & 7) * 8;
      async_b128(lds_addr(&sB[buf][r * 64 + c]),
                 &fc1[((size_t)(e * 512 + nb + r)) * 1024 + kb + c]);
    }
  };

  // phase 1: sH[128,512] = silu(hn2[tok] @ fc1[e]^T)
  for (int nb = 0; nb < 512; nb += 64) {
    v8f acc[2][2] = {};
    stage1(0, nb, 0);
    for (int s = 0; s < 16; ++s) {
      int cur = s & 1;
      if (s + 1 < 16) { stage1(cur ^ 1, nb, (s + 1) * 64); wait_async<6>(); }
      else wait_async<0>();
      __syncthreads();
#pragma unroll
      for (int ks = 0; ks < 2; ++ks) {
        v16bf af[2], bfr[2];
#pragma unroll
        for (int mi = 0; mi < 2; ++mi) af[mi] = load_a_frag(sA[cur], 64, wm + mi * 16, ks * 32);
#pragma unroll
        for (int ni = 0; ni < 2; ++ni) bfr[ni] = load_b_frag(sB[cur], 64, wn + ni * 16, ks * 32);
#pragma unroll
        for (int mi = 0; mi < 2; ++mi)
#pragma unroll
          for (int ni = 0; ni < 2; ++ni) acc[mi][ni] = wmma_bf16(af[mi], bfr[ni], acc[mi][ni]);
      }
      __syncthreads();
    }
#pragma unroll
    for (int mi = 0; mi < 2; ++mi)
#pragma unroll
      for (int ni = 0; ni < 2; ++ni)
#pragma unroll
        for (int r = 0; r < 8; ++r) {
          int row = wm + mi * 16 + half * 8 + r;
          int col = nb + wn + ni * 16 + nlo;
          float a = acc[mi][ni][r];
          sH[row * 512 + col] = f2bf(a * fast_rcp(1.f + __expf(-a)));
        }
  }
  __syncthreads();

  auto stage2 = [&](int buf, int nb, int kb) {
#pragma unroll
    for (int i = 0; i < 2; ++i) {
      int idx = tid + i * 256;
      int r = idx >> 3, c = (idx & 7) * 8;
      async_b128(lds_addr(&sB[buf][r * 64 + c]),
                 &fc2[((size_t)(e * 1024 + nb + r)) * 512 + kb + c]);
    }
  };

  // phase 2: out[tok] += wgt * (sH @ fc2[e]^T)
  for (int nb = 0; nb < 1024; nb += 64) {
    v8f acc[2][2] = {};
    stage2(0, nb, 0);
    for (int s = 0; s < 8; ++s) {
      int cur = s & 1;
      if (s + 1 < 8) { stage2(cur ^ 1, nb, (s + 1) * 64); wait_async<2>(); }
      else wait_async<0>();
      __syncthreads();
#pragma unroll
      for (int ks = 0; ks < 2; ++ks) {
        v16bf af[2], bfr[2];
#pragma unroll
        for (int mi = 0; mi < 2; ++mi)
          af[mi] = load_a_frag(sH, 512, wm + mi * 16, s * 64 + ks * 32);
#pragma unroll
        for (int ni = 0; ni < 2; ++ni) bfr[ni] = load_b_frag(sB[cur], 64, wn + ni * 16, ks * 32);
#pragma unroll
        for (int mi = 0; mi < 2; ++mi)
#pragma unroll
          for (int ni = 0; ni < 2; ++ni) acc[mi][ni] = wmma_bf16(af[mi], bfr[ni], acc[mi][ni]);
      }
      __syncthreads();
    }
#pragma unroll
    for (int mi = 0; mi < 2; ++mi)
#pragma unroll
      for (int ni = 0; ni < 2; ++ni)
#pragma unroll
        for (int r = 0; r < 8; ++r) {
          int row = wm + mi * 16 + half * 8 + r;
          int slot = tile * 128 + row;
          if (slot < cnt) {
            int col = nb + wn + ni * 16 + nlo;
            atomicAdd(&out[(size_t)sTok[row] * 1024 + col], acc[mi][ni][r] * sWgt[row]);
          }
        }
  }
}

// =============================== host side ===================================
extern "C" void kernel_launch(void* const* d_in, const int* in_sizes, int n_in,
                              void* d_out, int out_size, void* d_ws, size_t ws_size,
                              hipStream_t stream) {
  const float* x   = (const float*)d_in[0];
  const float* wq  = (const float*)d_in[2];
  const float* wk  = (const float*)d_in[3];
  const float* wv  = (const float*)d_in[4];
  const float* wo  = (const float*)d_in[5];
  const float* n1  = (const float*)d_in[6];
  const float* n2  = (const float*)d_in[7];
  const float* gw  = (const float*)d_in[8];
  const float* fc1 = (const float*)d_in[9];
  const float* fc2 = (const float*)d_in[10];
  const float* s1  = (const float*)d_in[11];
  const float* s2  = (const float*)d_in[12];
  const float* s3  = (const float*)d_in[13];
  float* out = (float*)d_out;

  const int T = 4096, D = 1024, S = 2048, H = 512, E = 16;
  char* p = (char*)d_ws;
  auto alloc = [&](size_t bytes) -> void* {
    void* r = (void*)p;
    p += (bytes + 255) & ~(size_t)255;
    return r;
  };
  bf16_t* wqb  = (bf16_t*)alloc((size_t)D * D * 2);
  bf16_t* wkb  = (bf16_t*)alloc((size_t)D * D * 2);
  bf16_t* wvb  = (bf16_t*)alloc((size_t)D * D * 2);
  bf16_t* wob  = (bf16_t*)alloc((size_t)D * D * 2);
  bf16_t* s1b  = (bf16_t*)alloc((size_t)S * D * 2);
  bf16_t* s2b  = (bf16_t*)alloc((size_t)S * D * 2);
  bf16_t* s3b  = (bf16_t*)alloc((size_t)D * S * 2);
  bf16_t* fc1b = (bf16_t*)alloc((size_t)E * H * D * 2);
  bf16_t* fc2b = (bf16_t*)alloc((size_t)E * D * H * 2);
  bf16_t* hnb  = (bf16_t*)alloc((size_t)T * D * 2);
  float*  qf32 = (float*) alloc((size_t)T * D * 4);
  float*  kf32 = (float*) alloc((size_t)T * D * 4);
  bf16_t* qb   = (bf16_t*)alloc((size_t)T * D * 2);
  bf16_t* kb   = (bf16_t*)alloc((size_t)T * D * 2);
  bf16_t* vb   = (bf16_t*)alloc((size_t)T * D * 2);
  bf16_t* attb = (bf16_t*)alloc((size_t)T * D * 2);
  float*  hf   = (float*) alloc((size_t)T * D * 4);
  bf16_t* hn2b = (bf16_t*)alloc((size_t)T * D * 2);
  bf16_t* gbuf = (bf16_t*)alloc((size_t)T * S * 2);
  int*    cnts = (int*)   alloc(64);
  int*    elist= (int*)   alloc((size_t)E * 4096 * 4);
  float*  ewt  = (float*) alloc((size_t)E * 4096 * 4);

  auto cvt = [&](const float* src, bf16_t* dst, int n) {
    cvt_bf16_kernel<<<(n + 255) / 256, 256, 0, stream>>>(src, dst, n);
  };
  cvt(wq, wqb, D * D);  cvt(wk, wkb, D * D);
  cvt(wv, wvb, D * D);  cvt(wo, wob, D * D);
  cvt(s1, s1b, S * D);  cvt(s2, s2b, S * D);  cvt(s3, s3b, D * S);
  cvt(fc1, fc1b, E * H * D);  cvt(fc2, fc2b, E * D * H);

  rmsnorm_kernel<<<T / 8, 256, 0, stream>>>(x, n1, hnb);

  dim3 gD(D / 64, T / 128);
  gemm_bf16_kernel<0><<<gD, 256, 0, stream>>>(hnb, wqb, nullptr, qf32, nullptr, D, D);
  gemm_bf16_kernel<0><<<gD, 256, 0, stream>>>(hnb, wkb, nullptr, kf32, nullptr, D, D);
  gemm_bf16_kernel<1><<<gD, 256, 0, stream>>>(hnb, wvb, nullptr, nullptr, vb, D, D);

  int rtot = T * 16 * 32;
  rope_kernel<<<(rtot + 255) / 256, 256, 0, stream>>>(qf32, qb, rtot);
  rope_kernel<<<(rtot + 255) / 256, 256, 0, stream>>>(kf32, kb, rtot);

  attn_kernel<<<dim3(1024 / 64, 4 * 16), 128, 0, stream>>>(qb, kb, vb, attb);

  gemm_bf16_kernel<2><<<gD, 256, 0, stream>>>(attb, wob, x, hf, nullptr, D, D);

  rmsnorm_kernel<<<T / 8, 256, 0, stream>>>(hf, n2, hn2b);

  gemm_silumul_kernel<<<dim3(S / 64, T / 128), 256, 0, stream>>>(hn2b, s1b, s2b, gbuf, S, D);
  // out = hf + shared   (MoE adds on top via atomics)
  gemm_bf16_kernel<2><<<gD, 256, 0, stream>>>(gbuf, s3b, hf, out, nullptr, D, S);

  zero_counts_kernel<<<1, 32, 0, stream>>>(cnts);
  gate_topk_kernel<<<T / 8, 256, 0, stream>>>(hn2b, gw, cnts, elist, ewt);
  moe_kernel<<<dim3(32, E), 256, 0, stream>>>(hn2b, fc1b, fc2b, cnts, elist, ewt, out);
}
